// CopyGenerator_37125697307272
// MI455X (gfx1250) — matrix operand order repaired and checked by reading, not compiled, not run on hardware
//
#include <hip/hip_runtime.h>
#include <hip/hip_bf16.h>
#include <math.h>

// Problem constants (from the reference)
#define B_DIM 32
#define T_DIM 128
#define S_LEN 400
#define HDIM  512
#define VDIM  32000
#define CEXT  620
#define OUTW  (VDIM + CEXT)    // 32620
#define NROWS (B_DIM * T_DIM)  // 4096

// GEMM blocking
#define M_BLK 128                          // rows per workgroup (8 WMMA row-tiles)
#define LDA   520                          // padded LDS row stride (bf16 elems):
                                           // 1040B = 260 dwords, 260%64=4 -> the 16
                                           // rows of a fragment load map to all 64
                                           // banks (conflict-free ds_load_b128)
#define NS    25                           // column groups (grid.x)
#define CHUNKS_PER_WG (VDIM / 128 / NS)    // 10 chunks of 128 cols each

typedef __bf16 bf16_t;
typedef __attribute__((ext_vector_type(16))) __bf16 v16bf;
typedef __attribute__((ext_vector_type(8)))  float  v8f;

union Frag  { v16bf v; float4 q[2]; };
union Pack4 { bf16_t e[4]; uint2 u; };

__device__ __forceinline__ void online_combine(float& m, float& s, float m2, float s2) {
  float mn = fmaxf(m, m2);
  s = s * __expf(m - mn) + s2 * __expf(m2 - mn);
  m = mn;
}

// ---------------------------------------------------------------------------
// Kprep: one-time f32 -> bf16 hi/lo split of W into workspace. The GEMM inner
// loop is then pure loads + WMMA; hi+lo bf16 carries ~16 mantissa bits.
// ---------------------------------------------------------------------------
__global__ __launch_bounds__(256) void prep_w(const float* __restrict__ W,
                                              bf16_t* __restrict__ Whi,
                                              bf16_t* __restrict__ Wlo) {
  const int idx = blockIdx.x * 256 + threadIdx.x;
  if (idx >= VDIM * HDIM / 4) return;
  float4 x = ((const float4*)W)[idx];
  float v[4] = {x.x, x.y, x.z, x.w};
  Pack4 h, l;
#pragma unroll
  for (int e = 0; e < 4; ++e) {
    bf16_t hb = (bf16_t)v[e];
    h.e[e] = hb;
    l.e[e] = (bf16_t)(v[e] - (float)hb);
  }
  ((uint2*)Whi)[idx] = h.u;
  ((uint2*)Wlo)[idx] = l.u;
}

// ---------------------------------------------------------------------------
// K0: p_copy[n] = sigmoid(hidden[n,:] . Wc + bc). One wave per row.
// ---------------------------------------------------------------------------
__global__ __launch_bounds__(256) void pcopy_kernel(
    const float* __restrict__ hidden, const float* __restrict__ Wc,
    const float* __restrict__ bc, float* __restrict__ pc) {
  const int wave = threadIdx.x >> 5;
  const int lane = threadIdx.x & 31;
  const int n = blockIdx.x * 8 + wave;
  float sum = 0.f;
  for (int k = lane; k < HDIM; k += 32)
    sum = fmaf(hidden[(size_t)n * HDIM + k], Wc[k], sum);
#pragma unroll
  for (int mask = 16; mask >= 1; mask >>= 1)
    sum += __shfl_xor(sum, mask, 32);
  if (lane == 0)
    pc[n] = 1.0f / (1.0f + __expf(-(sum + bc[0])));
}

// ---------------------------------------------------------------------------
// K1: raw logits = hidden @ W.T + b via bf16 hi/lo split WMMA.
// Workgroup = 128 rows x 128 cols per chunk. Wave grid is 2x4: each wave owns
// 4 row-tiles (wr half) x 2 col-tiles (wc quarter), so per k-step:
//   16 ds_load_b128 (A hi/lo, bank-conflict-free via LDA=520 padding)
//   8  global_load_b128 (B hi/lo from L2-resident pre-split W)
//   24 v_wmma_f32_16x16x32_bf16
// All A LDS offsets fit the 16-bit ds immediate (wave base + rti*16*LDA).
// ---------------------------------------------------------------------------
__global__ __launch_bounds__(256) void gemm_logits(
    const float* __restrict__ hidden, const bf16_t* __restrict__ Whi,
    const bf16_t* __restrict__ Wlo, const float* __restrict__ bias,
    float* __restrict__ out) {
  extern __shared__ char smem[];
  bf16_t* aHi = (bf16_t*)smem;
  bf16_t* aLo = aHi + M_BLK * LDA;

  const int tid = threadIdx.x;
  const int wave = tid >> 5;
  const int lane = tid & 31;
  const int rowbase = blockIdx.y * M_BLK;

  // Stage A panel into LDS (padded rows), split into bf16 hi + lo.
  for (int idx = tid; idx < M_BLK * HDIM; idx += 256) {
    const int r = idx >> 9;          // / HDIM
    const int k = idx & (HDIM - 1);
    float x = hidden[(size_t)(rowbase + r) * HDIM + k];
    bf16_t h = (bf16_t)x;
    aHi[r * LDA + k] = h;
    aLo[r * LDA + k] = (bf16_t)(x - (float)h);
  }
  __syncthreads();

  const int halfsel = lane >> 4;     // 0: lanes 0-15, 1: lanes 16-31
  const int l15     = lane & 15;
  const int akoff   = halfsel * 16;  // A frag upper half: K 16..31
  const int bkoff   = halfsel * 16;  // B frag upper half: K 16..31
  const int wr      = wave >> 2;     // row half (4 row-tiles)
  const int wc      = wave & 3;      // col quarter (2 col-tiles)

  const bf16_t* aHiW = aHi + (wr * 64 + l15) * LDA + akoff;
  const bf16_t* aLoW = aLo + (wr * 64 + l15) * LDA + akoff;

  for (int ch = 0; ch < CHUNKS_PER_WG; ++ch) {
    const int colbase = (blockIdx.x * CHUNKS_PER_WG + ch) * 128 + wc * 32 + l15;
    const bf16_t* whiP0 = Whi + (size_t)colbase * HDIM + bkoff;
    const bf16_t* wloP0 = Wlo + (size_t)colbase * HDIM + bkoff;
    const bf16_t* whiP1 = whiP0 + 16 * HDIM;
    const bf16_t* wloP1 = wloP0 + 16 * HDIM;

    v8f c[4][2];
    v8f z = {};
#pragma unroll
    for (int rt = 0; rt < 4; ++rt) { c[rt][0] = z; c[rt][1] = z; }

#pragma unroll 2
    for (int ks = 0; ks < HDIM / 32; ++ks) {
      const int kb = ks * 32;
      Frag bhi0, blo0, bhi1, blo1;  // B[k][col] = W[col][k]
      bhi0.q[0] = *(const float4*)(whiP0 + kb);
      bhi0.q[1] = *(const float4*)(whiP0 + kb + 8);
      blo0.q[0] = *(const float4*)(wloP0 + kb);
      blo0.q[1] = *(const float4*)(wloP0 + kb + 8);
      bhi1.q[0] = *(const float4*)(whiP1 + kb);
      bhi1.q[1] = *(const float4*)(whiP1 + kb + 8);
      blo1.q[0] = *(const float4*)(wloP1 + kb);
      blo1.q[1] = *(const float4*)(wloP1 + kb + 8);
#pragma unroll
      for (int rti = 0; rti < 4; ++rti) {
        const bf16_t* pHi = aHiW + rti * 16 * LDA + kb;
        const bf16_t* pLo = aLoW + rti * 16 * LDA + kb;
        Frag ahi, alo;
        ahi.q[0] = *(const float4*)pHi;
        ahi.q[1] = *(const float4*)(pHi + 8);
        alo.q[0] = *(const float4*)pLo;
        alo.q[1] = *(const float4*)(pLo + 8);
        c[rti][0] = __builtin_amdgcn_wmma_f32_16x16x32_bf16(false, ahi.v, false, bhi0.v, (short)0, c[rti][0], false, false);
        c[rti][0] = __builtin_amdgcn_wmma_f32_16x16x32_bf16(false, ahi.v, false, blo0.v, (short)0, c[rti][0], false, false);
        c[rti][0] = __builtin_amdgcn_wmma_f32_16x16x32_bf16(false, alo.v, false, bhi0.v, (short)0, c[rti][0], false, false);
        c[rti][1] = __builtin_amdgcn_wmma_f32_16x16x32_bf16(false, ahi.v, false, bhi1.v, (short)0, c[rti][1], false, false);
        c[rti][1] = __builtin_amdgcn_wmma_f32_16x16x32_bf16(false, ahi.v, false, blo1.v, (short)0, c[rti][1], false, false);
        c[rti][1] = __builtin_amdgcn_wmma_f32_16x16x32_bf16(false, alo.v, false, bhi1.v, (short)0, c[rti][1], false, false);
      }
    }

    const float bb0 = bias[colbase];
    const float bb1 = bias[colbase + 16];
#pragma unroll
    for (int rti = 0; rti < 4; ++rti) {
#pragma unroll
      for (int i = 0; i < 8; ++i) {
        const int mrow = wr * 64 + rti * 16 + i + halfsel * 8;  // C layout: VGPR i -> rows i / i+8
        float* orow = out + (size_t)(rowbase + mrow) * OUTW;
        orow[colbase]      = c[rti][0][i] + bb0;
        orow[colbase + 16] = c[rti][1][i] + bb1;
      }
    }
  }
}

// ---------------------------------------------------------------------------
// K2: fused softmax reduce + normalize. One workgroup per row: online
// (max,sumexp) per thread -> shfl_xor wave reduce -> cross-wave LDS combine,
// then out[n,v] = exp(logit - M) * (1 - p_copy[n]) / S. Second read of the
// 128KB row hits L2.
// ---------------------------------------------------------------------------
__global__ __launch_bounds__(256) void softmax_norm(float* __restrict__ out,
                                                    const float* __restrict__ pc) {
  const int n = blockIdx.x;
  float4* row4 = (float4*)(out + (size_t)n * OUTW);
  const int tid = threadIdx.x;
  const int wave = tid >> 5;
  const int lane = tid & 31;

  float m = -INFINITY, s = 0.f;
  for (int j = tid; j < VDIM / 4; j += 256) {
    float4 x = row4[j];
    float vv[4] = {x.x, x.y, x.z, x.w};
#pragma unroll
    for (int e = 0; e < 4; ++e) {
      float mn = fmaxf(m, vv[e]);
      s = s * __expf(m - mn) + __expf(vv[e] - mn);
      m = mn;
    }
  }
#pragma unroll
  for (int mask = 1; mask <= 16; mask <<= 1) {
    float m2 = __shfl_xor(m, mask, 32);
    float s2 = __shfl_xor(s, mask, 32);
    online_combine(m, s, m2, s2);
  }

  __shared__ float wm[8], wsv[8], bM[1], bSc[1];
  if (lane == 0) { wm[wave] = m; wsv[wave] = s; }
  __syncthreads();
  if (tid == 0) {
    float M = wm[0], S = wsv[0];
#pragma unroll
    for (int w = 1; w < 8; ++w) online_combine(M, S, wm[w], wsv[w]);
    bM[0] = M;
    bSc[0] = (1.0f - pc[n]) / S;
  }
  __syncthreads();
  const float M = bM[0], sc = bSc[0];

  for (int j = tid; j < VDIM / 4; j += 256) {
    float4 x = row4[j];
    x.x = __expf(x.x - M) * sc;
    x.y = __expf(x.y - M) * sc;
    x.z = __expf(x.z - M) * sc;
    x.w = __expf(x.w - M) * sc;
    row4[j] = x;
  }
}

// ---------------------------------------------------------------------------
// K3: copy_ext[t,b,c] = p_copy * sum_s attn[t*B+b, s] * src_map[s,b,c].
// Deterministic LDS-tiled dense reduction (no float atomics).
// ---------------------------------------------------------------------------
__global__ __launch_bounds__(256) void copy_ext_kernel(
    const float* __restrict__ attn, const float* __restrict__ src_map,
    const float* __restrict__ pc, float* __restrict__ out) {
  __shared__ float cp[16][S_LEN];
  __shared__ float srows[16][624];  // padded to 39*16, zero-filled past 620

  const int tid = threadIdx.x;
  const int b = blockIdx.y;
  const int tbase = blockIdx.x * 16;
  const int t_local = tid >> 4;
  const int clane = tid & 15;

  for (int t = 0; t < 16; ++t) {
    const int n = (tbase + t) * B_DIM + b;
    const float g = pc[n];
    for (int sidx = tid; sidx < S_LEN; sidx += 256)
      cp[t][sidx] = attn[(size_t)n * S_LEN + sidx] * g;
  }

  float acc[39];
#pragma unroll
  for (int k = 0; k < 39; ++k) acc[k] = 0.f;

  for (int sb = 0; sb < S_LEN / 16; ++sb) {
    __syncthreads();
    for (int j = 0; j < 16; ++j) {
      const int s = sb * 16 + j;
      const float* srcrow = src_map + ((size_t)s * B_DIM + b) * CEXT;
      for (int cc = tid; cc < 624; cc += 256)
        srows[j][cc] = (cc < CEXT) ? srcrow[cc] : 0.f;
    }
    __syncthreads();
#pragma unroll 2
    for (int j = 0; j < 16; ++j) {
      const float a = cp[t_local][sb * 16 + j];
#pragma unroll
      for (int k = 0; k < 39; ++k)
        acc[k] = fmaf(a, srows[j][clane + 16 * k], acc[k]);
    }
  }

  const int n = (tbase + t_local) * B_DIM + b;
  float* orow = out + (size_t)n * OUTW + VDIM;
#pragma unroll
  for (int k = 0; k < 39; ++k) {
    const int c = clane + 16 * k;
    if (c < CEXT) orow[c] = acc[k];
  }
}

// ---------------------------------------------------------------------------
// Launch. Inputs (all f32): hidden, attn, W, b, Wc, bc, src_map.
// Workspace: pc[4096] f32 | Whi[V*H] bf16 | Wlo[V*H] bf16  (~65.6 MB).
// ---------------------------------------------------------------------------
extern "C" void kernel_launch(void* const* d_in, const int* in_sizes, int n_in,
                              void* d_out, int out_size, void* d_ws, size_t ws_size,
                              hipStream_t stream) {
  const float* hidden  = (const float*)d_in[0];
  const float* attn    = (const float*)d_in[1];
  const float* W       = (const float*)d_in[2];
  const float* bias    = (const float*)d_in[3];
  const float* Wc      = (const float*)d_in[4];
  const float* bc      = (const float*)d_in[5];
  const float* src_map = (const float*)d_in[6];
  float* out = (float*)d_out;

  float*  pc  = (float*)d_ws;
  bf16_t* Whi = (bf16_t*)(pc + NROWS);
  bf16_t* Wlo = Whi + (size_t)VDIM * HDIM;

  // Kprep: one-time hi/lo bf16 split of W
  prep_w<<<(VDIM * HDIM / 4 + 255) / 256, 256, 0, stream>>>(W, Whi, Wlo);

  // K0: copy gate
  pcopy_kernel<<<NROWS / 8, 256, 0, stream>>>(hidden, Wc, bc, pc);

  // K1: WMMA GEMM -> raw logits (+bias) into d_out
  const size_t lds_bytes = 2 * (size_t)M_BLK * LDA * sizeof(bf16_t);  // 260 KB
  gemm_logits<<<dim3(NS, NROWS / M_BLK), 256, lds_bytes, stream>>>(
      hidden, Whi, Wlo, bias, out);

  // K2: fused softmax reduce + (1-p_copy) scaling
  softmax_norm<<<NROWS, 256, 0, stream>>>(out, pc);

  // K3: copy distribution scattered into extended vocab
  copy_ext_kernel<<<dim3(T_DIM / 16, B_DIM), 256, 0, stream>>>(attn, src_map,
                                                               pc, out);
}
